// MultiHeadAttention_44315472560180
// MI455X (gfx1250) — matrix operand order repaired and checked by reading, not compiled, not run on hardware
//
#include <hip/hip_runtime.h>
#include <hip/hip_bf16.h>
#include <math.h>

// MHA forward for MI455X (gfx1250, wave32, WMMA + TDM + async-LDS).
// B=2, S=2048, E=1024, H=16, D=64. All GEMMs via v_wmma_f32_16x16x32_bf16.

typedef __attribute__((ext_vector_type(16))) __bf16 v16bf;
typedef __attribute__((ext_vector_type(8)))  float  v8f;
typedef __attribute__((ext_vector_type(4)))  unsigned int u32x4;
typedef __attribute__((ext_vector_type(8)))  unsigned int u32x8;

union Frag16 { v16bf v; uint4 q[2]; };
union F8     { v8f v; float f[8]; };
union H8     { uint4 q; __bf16 h[8]; };

__device__ __forceinline__ v8f wmma_bf16(v16bf a, v16bf b, v8f c) {
  // (neg_a, A, neg_b, B, c_mod, C, reuse_a, reuse_b)
  return __builtin_amdgcn_wmma_f32_16x16x32_bf16(false, a, false, b, (short)0, c,
                                                 false, false);
}

__device__ __forceinline__ unsigned int packbf2(float a, float b) {
  union { __bf16 h[2]; unsigned int u; } t;
  t.h[0] = (__bf16)a; t.h[1] = (__bf16)b;
  return t.u;
}

// 16x32 bf16 A fragment (ISA 7.12.2): lanes 0-15 row M=lane, K = {0..7,16..23};
// lanes 16-31 row M=lane-16, K = {8..15,24..31}. Rows are PITCH halves apart.
template<int PITCH>
__device__ __forceinline__ v16bf load_a_frag(const __bf16* base, int lane) {
  const __bf16* p = base + (lane & 15) * PITCH + ((lane < 16) ? 0 : 8);
  Frag16 f;
  f.q[0] = *(const uint4*)(p);
  f.q[1] = *(const uint4*)(p + 16);
  return f.v;
}

// 32x16 bf16 B fragment read from K-transposed storage: row index = output col N,
// PITCH halves over K. Lanes 0-15 hold K={0..7,16..23} of col N=lane, lanes 16-31 K={8..15,24..31}.
template<int PITCH>
__device__ __forceinline__ v16bf load_b_frag(const __bf16* base, int lane) {
  const __bf16* p = base + (lane & 15) * PITCH + ((lane < 16) ? 0 : 8);
  Frag16 f;
  f.q[0] = *(const uint4*)(p);
  f.q[1] = *(const uint4*)(p + 16);
  return f.v;
}

__device__ __forceinline__ float half_max(float v) {
  v = fmaxf(v, __shfl_xor(v, 1));
  v = fmaxf(v, __shfl_xor(v, 2));
  v = fmaxf(v, __shfl_xor(v, 4));
  v = fmaxf(v, __shfl_xor(v, 8));
  return v;
}
__device__ __forceinline__ float half_sum(float v) {
  v += __shfl_xor(v, 1);
  v += __shfl_xor(v, 2);
  v += __shfl_xor(v, 4);
  v += __shfl_xor(v, 8);
  return v;
}

// ---------------------------------------------------------------------------
// GEMM: C[4096,1024] = relu(X[4096,1024] @ W[1024,1024] + bias)
// Block tile 128x128, 8 waves, each wave 32x64 (2x4 WMMA tiles), k-step 32.
// IN_BF16: X is bf16 (attention output), staged via global_load_async_to_lds.
// HEAD_MAJOR_OUT: write bf16 to [B,H,S,D] workspace; else fp32 to [M,N].
// ---------------------------------------------------------------------------
template<bool IN_BF16, bool HEAD_MAJOR_OUT>
__global__ __launch_bounds__(256)
void gemm_bias_relu(const void* __restrict__ Xv, const float* __restrict__ W,
                    const float* __restrict__ bias, void* __restrict__ Outv) {
  // Manual LDS partition so byte offsets are known for async-LDS ops.
  __shared__ __align__(16) __bf16 smem[8192];           // 16 KB
  __bf16 (*Abuf)[32] = (__bf16(*)[32])(smem);           // LDS byte offset 0
  __bf16 (*Btl)[32]  = (__bf16(*)[32])(smem + 4096);    // W transposed [n][k]

  const int tid = threadIdx.x;
  const int lane = tid & 31, wave = tid >> 5;
  const int m0 = blockIdx.y * 128, n0 = blockIdx.x * 128;
  const int wm = (wave >> 1) * 32, wn = (wave & 1) * 64;

  F8 zf;
  #pragma unroll
  for (int i = 0; i < 8; ++i) zf.f[i] = 0.f;
  v8f acc[2][4];
  #pragma unroll
  for (int s = 0; s < 2; ++s)
    #pragma unroll
    for (int t = 0; t < 4; ++t) acc[s][t] = zf.v;

  for (int k0 = 0; k0 < 1024; k0 += 32) {
    // stage A (128 rows x 32 k) as bf16
    if (IN_BF16) {
      // straight bf16 copy -> async LDS path (ASYNCcnt), no VGPR bounce
      const __bf16* X = (const __bf16*)Xv;
      #pragma unroll
      for (int p = 0; p < 2; ++p) {
        int row = (tid >> 2) + p * 64;
        int c8  = (tid & 3) * 8;
        unsigned lds_off = (unsigned)(row * 64 + c8 * 2);   // bytes, Abuf @ 0
        unsigned long long ga =
            (unsigned long long)(X + (size_t)(m0 + row) * 1024 + k0 + c8);
        asm volatile("global_load_async_to_lds_b128 %0, %1, off"
                     :: "v"(lds_off), "v"(ga) : "memory");
      }
    } else {
      const float* X = (const float*)Xv;
      #pragma unroll
      for (int p = 0; p < 4; ++p) {
        int row = (tid >> 3) + p * 32;
        int c4  = (tid & 7) * 4;
        float4 v = *(const float4*)(X + (size_t)(m0 + row) * 1024 + k0 + c4);
        unsigned int* d = (unsigned int*)&Abuf[row][c4];
        d[0] = packbf2(v.x, v.y);
        d[1] = packbf2(v.z, v.w);
      }
    }
    // stage W tile transposed (32 k x 128 n) -> Btl[n][k], fp32 -> bf16
    #pragma unroll
    for (int p = 0; p < 4; ++p) {
      int kk = (tid >> 5) + p * 8;
      int n4 = (tid & 31) * 4;
      float4 v = *(const float4*)(W + (size_t)(k0 + kk) * 1024 + n0 + n4);
      Btl[n4 + 0][kk] = (__bf16)v.x;
      Btl[n4 + 1][kk] = (__bf16)v.y;
      Btl[n4 + 2][kk] = (__bf16)v.z;
      Btl[n4 + 3][kk] = (__bf16)v.w;
      if (k0 + 32 < 1024)
        __builtin_prefetch(W + (size_t)(k0 + 32 + kk) * 1024 + n0 + n4, 0, 1);
    }
    if (IN_BF16)
      asm volatile("s_wait_asynccnt 0x0" ::: "memory");
    __syncthreads();

    v16bf af0 = load_a_frag<32>(&Abuf[wm][0], lane);
    v16bf af1 = load_a_frag<32>(&Abuf[wm + 16][0], lane);
    #pragma unroll
    for (int t = 0; t < 4; ++t) {
      v16bf bfr = load_b_frag<32>(&Btl[wn + t * 16][0], lane);
      acc[0][t] = wmma_bf16(af0, bfr, acc[0][t]);
      acc[1][t] = wmma_bf16(af1, bfr, acc[1][t]);
    }
    __syncthreads();
  }

  // epilogue: bias + relu, store. C layout: VGPR r -> (M = r + 8*(lane/16), N = lane%16)
  const int hi8 = (lane >= 16) ? 8 : 0;
  #pragma unroll
  for (int t = 0; t < 4; ++t) {
    const int n = n0 + wn + t * 16 + (lane & 15);
    const float bv = bias[n];
    #pragma unroll
    for (int s = 0; s < 2; ++s) {
      F8 c; c.v = acc[s][t];
      #pragma unroll
      for (int r = 0; r < 8; ++r) {
        const int m = m0 + wm + s * 16 + r + hi8;
        float val = fmaxf(c.f[r] + bv, 0.f);
        if (HEAD_MAJOR_OUT) {
          // [B,H,S,D]: b = m/2048, s = m%2048, h = n/64, d = n%64
          const int bb = m >> 11, sq = m & 2047, hh = n >> 6, dd = n & 63;
          ((__bf16*)Outv)[(((size_t)(bb * 16 + hh) * 2048) + sq) * 64 + dd] =
              (__bf16)val;
        } else {
          ((float*)Outv)[(size_t)m * 1024 + n] = val;
        }
      }
    }
  }
}

// ---------------------------------------------------------------------------
// Flash attention: grid (S/128 q-tiles, B*H). 8 waves, each wave = 16 q rows.
// qh/kh/vh bf16 [B,H,S,D]; y bf16 [B,S,H,D] (== [M,E] for the out projection).
// K tile staged by the Tensor Data Mover (tensor_load_to_lds, TENSORcnt).
// ---------------------------------------------------------------------------
__global__ __launch_bounds__(256)
void attention_kernel(const __bf16* __restrict__ qh, const __bf16* __restrict__ kh,
                      const __bf16* __restrict__ vh, __bf16* __restrict__ y) {
  // LDS partition (byte offsets): Ksh @ 0 (8KB), Vt @ 8192 (8KB), P @ 16384 (16KB)
  __shared__ __align__(16) __bf16 smem[16384];          // 32 KB
  __bf16 (*Ksh)[64] = (__bf16(*)[64])(smem);            // [key][d]
  __bf16 (*Vt)[64]  = (__bf16(*)[64])(smem + 4096);     // [d][key]

  const int tid = threadIdx.x, lane = tid & 31, wave = tid >> 5;
  const int qt = blockIdx.x, bh = blockIdx.y;
  const int b = bh >> 4, h = bh & 15;
  const int q0 = qt * 128 + wave * 16;
  const size_t baseBH = (size_t)bh * 2048 * 64;
  const int hi8 = (lane >= 16) ? 8 : 0;
  __bf16* Pw = smem + 8192 + wave * 16 * 64;            // per-wave P tile [16][64]

  // preload Q fragments (A layout) straight from global
  const v16bf qf0 = load_a_frag<64>(qh + baseBH + (size_t)q0 * 64 + 0, lane);
  const v16bf qf1 = load_a_frag<64>(qh + baseBH + (size_t)q0 * 64 + 32, lane);

  F8 zf;
  #pragma unroll
  for (int i = 0; i < 8; ++i) zf.f[i] = 0.f;
  v8f o[4];
  #pragma unroll
  for (int ds = 0; ds < 4; ++ds) o[ds] = zf.v;
  float m[8], ls[8];
  #pragma unroll
  for (int r = 0; r < 8; ++r) { m[r] = -1e30f; ls[r] = 0.f; }

  const int kbmax = (qt * 128 + 127) >> 6;   // causal: key blocks 0..kbmax
  for (int kb = 0; kb <= kbmax; ++kb) {
    const int key0 = kb * 64;
    __syncthreads();

    // --- K tile: one TDM descriptor, issued by wave 0 only (EXEC ignored).
    // D# per ISA 8.3/8.4: 2D tile 64(d) x 64(key), bf16, dense copy into Ksh@0.
    if (wave == 0) {
      unsigned long long ga =
          (unsigned long long)(kh + baseBH + (size_t)key0 * 64);
      u32x4 g0;
      g0[0] = 1u;                                   // count=1, user descriptor
      g0[1] = 0u;                                   // lds_addr = 0 (Ksh)
      g0[2] = (unsigned)(ga & 0xffffffffu);         // global_addr[31:0]
      g0[3] = (unsigned)((ga >> 32) & 0x01ffffffu)  // global_addr[56:32]
              | 0x80000000u;                        // type = 2 ("image")
      u32x8 g1;
      g1[0] = (1u << 16);        // data_size = 1 (2 bytes/elem)
      g1[1] = (64u << 16);       // tensor_dim0 = 64  (bits 79:48, low half)
      g1[2] = (2048u << 16);     // tensor_dim0 hi = 0 | tensor_dim1 = 2048 lo
      g1[3] = (64u << 16);       // tensor_dim1 hi = 0 | tile_dim0 = 64
      g1[4] = 64u;               // tile_dim1 = 64, tile_dim2 = 0
      g1[5] = 64u;               // tensor_dim0_stride = 64 (low 32)
      g1[6] = 0u;                // stride0 hi | tensor_dim1_stride lo (unused 2D)
      g1[7] = 0u;
      asm volatile("tensor_load_to_lds %0, %1" :: "s"(g0), "s"(g1) : "memory");
    }

    // --- V tile: fp-transpose into Vt[d][key] (TDM cannot transpose)
    #pragma unroll
    for (int p = 0; p < 2; ++p) {
      const int c = tid + p * 256;        // 512 chunks of 8 halves
      const int key = c >> 3, c8 = (c & 7) * 8;
      H8 t;
      t.q = *(const uint4*)(vh + baseBH + (size_t)(key0 + key) * 64 + c8);
      #pragma unroll
      for (int i = 0; i < 8; ++i) Vt[c8 + i][key] = t.h[i];
    }
    if (wave == 0) __builtin_amdgcn_s_wait_tensorcnt(0);
    __syncthreads();

    // scores: 4 subtiles of 16 keys, K-dim = d (2 steps of 32)
    F8 sf[4];
    #pragma unroll
    for (int sub = 0; sub < 4; ++sub) {
      v8f acc = zf.v;
      acc = wmma_bf16(qf0, load_b_frag<64>(&Ksh[sub * 16][0], lane), acc);
      acc = wmma_bf16(qf1, load_b_frag<64>(&Ksh[sub * 16][32], lane), acc);
      sf[sub].v = acc;
      const int key = key0 + sub * 16 + (lane & 15);
      #pragma unroll
      for (int r = 0; r < 8; ++r) {
        const int qrow = q0 + r + hi8;
        const float vv = sf[sub].f[r] * 0.125f;   // 1/sqrt(64)
        sf[sub].f[r] = (key > qrow) ? -1e30f : vv;
      }
    }

    // online softmax: rows live per (VGPR r, half-wave); reduce across 16 lanes
    float scl_arr[8];
    #pragma unroll
    for (int r = 0; r < 8; ++r) {
      float lm = fmaxf(fmaxf(sf[0].f[r], sf[1].f[r]),
                       fmaxf(sf[2].f[r], sf[3].f[r]));
      lm = half_max(lm);
      const float mnew = fmaxf(m[r], lm);
      const float scl = __expf(m[r] - mnew);
      float psum = 0.f;
      #pragma unroll
      for (int sub = 0; sub < 4; ++sub) {
        const float p = __expf(sf[sub].f[r] - mnew);
        sf[sub].f[r] = p;
        psum += p;
      }
      psum = half_sum(psum);
      ls[r] = ls[r] * scl + psum;
      m[r] = mnew;
      scl_arr[r] = scl;
    }
    // rescale O accumulators
    #pragma unroll
    for (int ds = 0; ds < 4; ++ds) {
      F8 t; t.v = o[ds];
      #pragma unroll
      for (int r = 0; r < 8; ++r) t.f[r] *= scl_arr[r];
      o[ds] = t.v;
    }
    // write P (C layout -> [row][key]) into per-wave LDS for A-fragment reuse
    #pragma unroll
    for (int sub = 0; sub < 4; ++sub) {
      const int col = sub * 16 + (lane & 15);
      #pragma unroll
      for (int r = 0; r < 8; ++r)
        Pw[(r + hi8) * 64 + col] = (__bf16)sf[sub].f[r];
    }
    // P @ V: K-dim = key (2 steps of 32), 4 d-subtiles
    const v16bf pa0 = load_a_frag<64>(Pw + 0, lane);
    const v16bf pa1 = load_a_frag<64>(Pw + 32, lane);
    #pragma unroll
    for (int ds = 0; ds < 4; ++ds) {
      v8f acc = o[ds];
      acc = wmma_bf16(pa0, load_b_frag<64>(&Vt[ds * 16][0], lane), acc);
      acc = wmma_bf16(pa1, load_b_frag<64>(&Vt[ds * 16][32], lane), acc);
      o[ds] = acc;
    }
  }

  // normalize and store y as [B,S,H,D]
  float inv[8];
  #pragma unroll
  for (int r = 0; r < 8; ++r) inv[r] = 1.f / ls[r];
  #pragma unroll
  for (int ds = 0; ds < 4; ++ds) {
    F8 t; t.v = o[ds];
    const int d = ds * 16 + (lane & 15);
    #pragma unroll
    for (int r = 0; r < 8; ++r) {
      const int qrow = q0 + r + hi8;
      y[(((size_t)(b * 2048 + qrow)) * 16 + h) * 64 + d] = (__bf16)(t.f[r] * inv[r]);
    }
  }
}

// ---------------------------------------------------------------------------
extern "C" void kernel_launch(void* const* d_in, const int* in_sizes, int n_in,
                              void* d_out, int out_size, void* d_ws, size_t ws_size,
                              hipStream_t stream) {
  (void)in_sizes; (void)n_in; (void)out_size; (void)ws_size;
  const float* q  = (const float*)d_in[0];
  const float* k  = (const float*)d_in[1];
  const float* v  = (const float*)d_in[2];
  // d_in[3] is the causal mask; causality implemented directly.
  const float* Wq = (const float*)d_in[4];
  const float* bq = (const float*)d_in[5];
  const float* Wk = (const float*)d_in[6];
  const float* bk = (const float*)d_in[7];
  const float* Wv = (const float*)d_in[8];
  const float* bv = (const float*)d_in[9];
  const float* Wo = (const float*)d_in[10];
  const float* bo = (const float*)d_in[11];

  const size_t NELEM = (size_t)4096 * 1024;   // B*S*E
  __bf16* wsb = (__bf16*)d_ws;
  __bf16* qh = wsb;                 // [B,H,S,D] bf16
  __bf16* kh = wsb + NELEM;
  __bf16* vh = wsb + 2 * NELEM;
  __bf16* y  = wsb + 3 * NELEM;     // [B,S,H,D] == [M,E] bf16

  const dim3 gproj(8, 32);          // N/128, M/128
  const dim3 blk(256);
  gemm_bias_relu<false, true><<<gproj, blk, 0, stream>>>(q, Wq, bq, qh);
  gemm_bias_relu<false, true><<<gproj, blk, 0, stream>>>(k, Wk, bk, kh);
  gemm_bias_relu<false, true><<<gproj, blk, 0, stream>>>(v, Wv, bv, vh);

  attention_kernel<<<dim3(16, 32), blk, 0, stream>>>(qh, kh, vh, y);

  gemm_bias_relu<true, false><<<gproj, blk, 0, stream>>>(y, Wo, bo, d_out);
}
